// MutliHeadAttention2D_44092134261206
// MI455X (gfx1250) — matrix operand order repaired and checked by reading, not compiled, not run on hardware
//
#include <hip/hip_runtime.h>

#define B_    8
#define CIN   64
#define CO    64
#define H_    128
#define W_    128
#define HW    (H_ * W_)
#define GROUPS 8
#define CG    (CO / GROUPS)   // 8 channels per group

typedef __attribute__((ext_vector_type(2)))  float  v2f;
typedef __attribute__((ext_vector_type(8)))  float  v8f;
typedef __attribute__((ext_vector_type(16))) __bf16 v16bf;

#if defined(__has_builtin)
#if __has_builtin(__builtin_amdgcn_wmma_f32_16x16x4_f32)
#define USE_WMMA_F32 1
#endif
#endif

// ---------------------------------------------------------------------------
// Kernel 1: q/k/v = W{q,k,v} (64x64) @ x (64 x B*HW), via WMMA.
// One wave handles one batch b and one 16-pixel tile n0: it loads the full
// 64-deep x column once into registers (B-operand), then runs 3 matrices x
// 4 M-tiles of 16x16 output, accumulating over K=64.
//
// ISA 7.12.2 operand layouts used (wave32):
//   A 16xK (32b): lane t -> row M = t%16;  VGPRs hold K = (t/16)*2 + {0,1} per chunk
//   B Kx16 (32b): lane t -> col N = t%16;  VGPRs hold K rows, lane-halves split K
//   D 16x16 f32:  lane t -> col N = t%16;  VGPR j -> row M = j + 8*(t/16)
// ---------------------------------------------------------------------------
__global__ __launch_bounds__(256) void qkv_proj_wmma(
    const float* __restrict__ x,
    const float* __restrict__ Wq, const float* __restrict__ Wk,
    const float* __restrict__ Wv,
    float* __restrict__ q, float* __restrict__ k, float* __restrict__ v)
{
    const int lane  = threadIdx.x & 31;
    const int wid   = (blockIdx.x * blockDim.x + threadIdx.x) >> 5; // global wave
    const int b     = wid >> 10;            // 1024 pixel-tiles per batch
    const int n0    = (wid & 1023) << 4;    // pixel-tile base within HW
    const int mrow  = lane & 15;            // M row / N col / pixel within tile
    const int khalf = lane >> 4;            // lane-half selects K slice

    const float* xb = x + (size_t)b * CIN * HW + n0 + mrow;

    const float* Ws[3]   = {Wq, Wk, Wv};
    float*       Outs[3] = {q, k, v};

#if USE_WMMA_F32
    // ---- exact fp32 path: V_WMMA_F32_16X16X4_F32, 16 K-chunks of 4 ----
    // B-operand column of x, kept in registers across all 12 output tiles.
    v2f xcol[16];
#pragma unroll
    for (int kc = 0; kc < 16; ++kc) {
        const int c = kc * 4 + khalf * 2;          // K rows held by this lane
        xcol[kc].x = xb[(size_t)(c + 0) * HW];
        xcol[kc].y = xb[(size_t)(c + 1) * HW];
    }
#pragma unroll
    for (int mat = 0; mat < 3; ++mat) {
        const float* Wm = Ws[mat];
        float*       Om = Outs[mat] + (size_t)b * CO * HW + n0 + mrow;
#pragma unroll
        for (int mt = 0; mt < 4; ++mt) {
            v8f acc = {};
            const float* wr = Wm + (mt * 16 + mrow) * CIN + khalf * 2;
#pragma unroll
            for (int kc = 0; kc < 16; ++kc) {
                v2f a;
                a.x = wr[kc * 4 + 0];
                a.y = wr[kc * 4 + 1];
                acc = __builtin_amdgcn_wmma_f32_16x16x4_f32(
                    false, a, false, xcol[kc], (short)0, acc, false, false);
            }
#pragma unroll
            for (int j = 0; j < 8; ++j) {
                const int m = mt * 16 + j + 8 * khalf;   // D row for this lane
                Om[(size_t)m * HW] = acc[j];
            }
        }
    }
#else
    // ---- probe-confirmed fallback: V_WMMA_F32_16X16X32_BF16, 2 K-chunks ----
    v16bf xcol[2];
#pragma unroll
    for (int ch = 0; ch < 2; ++ch) {
#pragma unroll
        for (int e = 0; e < 16; ++e) {
            const int c = ch * 32 + khalf * 16 + e;     // B: lane-half splits K
            xcol[ch][e] = (__bf16)xb[(size_t)c * HW];
        }
    }
#pragma unroll
    for (int mat = 0; mat < 3; ++mat) {
        const float* Wm = Ws[mat];
        float*       Om = Outs[mat] + (size_t)b * CO * HW + n0 + mrow;
#pragma unroll
        for (int mt = 0; mt < 4; ++mt) {
            v8f acc = {};
#pragma unroll
            for (int ch = 0; ch < 2; ++ch) {
                v16bf a;
#pragma unroll
                for (int e = 0; e < 16; ++e) {
                    // A layout: element e holds K = khalf*16 + e (matches B)
                    const int kk = khalf * 16 + e;
                    a[e] = (__bf16)Wm[(mt * 16 + mrow) * CIN + ch * 32 + kk];
                }
                acc = __builtin_amdgcn_wmma_f32_16x16x32_bf16(
                    false, a, false, xcol[ch], (short)0, acc, false, false);
            }
#pragma unroll
            for (int j = 0; j < 8; ++j) {
                const int m = mt * 16 + j + 8 * khalf;
                Om[(size_t)m * HW] = acc[j];
            }
        }
    }
#endif
}

// ---------------------------------------------------------------------------
// Kernel 2: local 3x3 grouped attention. One thread per (b, g, h, w).
// energy = dot(q_g, k_g@(h+dt-1,w+df-1)) + (g<4 ? tbias[dt] : fbias[df]);
// k/v are implicitly zero-padded (bias still applies at padded positions,
// matching the reference which adds bias after unfolding padded k).
// ---------------------------------------------------------------------------
__global__ __launch_bounds__(256) void local_attn_kernel(
    const float* __restrict__ q, const float* __restrict__ k,
    const float* __restrict__ v,
    const float* __restrict__ rel_t, const float* __restrict__ rel_f,
    float* __restrict__ out, float* __restrict__ attn)
{
    const int tid = blockIdx.x * 256 + threadIdx.x;   // B*G*H*W threads
    const int w = tid & (W_ - 1);
    const int h = (tid >> 7) & (H_ - 1);
    const int g = (tid >> 14) & (GROUPS - 1);
    const int b = tid >> 17;

    const size_t cbase = ((size_t)b * CO + g * CG) * HW;  // channel base
    const size_t pix   = (size_t)h * W_ + w;

    float qv[CG];
#pragma unroll
    for (int j = 0; j < CG; ++j) qv[j] = q[cbase + (size_t)j * HW + pix];

    // bias3[dt] (groups 0-3, time) or bias3[df] (groups 4-7, freq)
    float bias3[3];
    if (g < 4) {
#pragma unroll
        for (int t = 0; t < 3; ++t) {
            float s = 0.f;
#pragma unroll
            for (int j = 0; j < CG; ++j) s = fmaf(qv[j], rel_t[(g * CG + j) * 3 + t], s);
            bias3[t] = s;
        }
    } else {
#pragma unroll
        for (int t = 0; t < 3; ++t) {
            float s = 0.f;
#pragma unroll
            for (int j = 0; j < CG; ++j) s = fmaf(qv[j], rel_f[((g - 4) * CG + j) * 3 + t], s);
            bias3[t] = s;
        }
    }

    float e[9];
    float mx = -3.0e38f;
#pragma unroll
    for (int dt = 0; dt < 3; ++dt) {
#pragma unroll
        for (int df = 0; df < 3; ++df) {
            const int hh = h + dt - 1, ww = w + df - 1;
            float dot = 0.f;
            if (hh >= 0 && hh < H_ && ww >= 0 && ww < W_) {
                const size_t kp = cbase + (size_t)hh * W_ + ww;
#pragma unroll
                for (int j = 0; j < CG; ++j) dot = fmaf(qv[j], k[kp + (size_t)j * HW], dot);
            }
            const float en = dot + (g < 4 ? bias3[dt] : bias3[df]);
            e[dt * 3 + df] = en;
            mx = fmaxf(mx, en);
        }
    }
    float s = 0.f;
#pragma unroll
    for (int i = 0; i < 9; ++i) { e[i] = __expf(e[i] - mx); s += e[i]; }
    const float inv = 1.f / s;

    float acc[CG];
#pragma unroll
    for (int j = 0; j < CG; ++j) acc[j] = 0.f;
#pragma unroll
    for (int dt = 0; dt < 3; ++dt) {
#pragma unroll
        for (int df = 0; df < 3; ++df) {
            const int hh = h + dt - 1, ww = w + df - 1;
            if (hh >= 0 && hh < H_ && ww >= 0 && ww < W_) {
                const float wgt = e[dt * 3 + df] * inv;
                const size_t vp = cbase + (size_t)hh * W_ + ww;
#pragma unroll
                for (int j = 0; j < CG; ++j)
                    acc[j] = fmaf(wgt, v[vp + (size_t)j * HW], acc[j]);
            }
        }
    }
#pragma unroll
    for (int j = 0; j < CG; ++j) out[cbase + (size_t)j * HW + pix] = acc[j];

    float* ap = attn + (size_t)tid * 9;   // [B,G,H,W,9], tid == ((b*G+g)*H+h)*W+w
#pragma unroll
    for (int i = 0; i < 9; ++i) ap[i] = e[i] * inv;
}

extern "C" void kernel_launch(void* const* d_in, const int* in_sizes, int n_in,
                              void* d_out, int out_size, void* d_ws, size_t ws_size,
                              hipStream_t stream) {
    const float* x     = (const float*)d_in[0];
    const float* Wq    = (const float*)d_in[1];
    const float* Wk    = (const float*)d_in[2];
    const float* Wv    = (const float*)d_in[3];
    const float* rel_t = (const float*)d_in[4];
    const float* rel_f = (const float*)d_in[5];

    const size_t plane = (size_t)B_ * CO * HW;   // 8,388,608 floats per tensor
    float* q = (float*)d_ws;
    float* k = q + plane;
    float* v = k + plane;

    float* out  = (float*)d_out;
    float* attn = out + plane;                   // attn follows out, flat

    // 8192 waves total = B * (HW/16) pixel-tiles; 8 waves per 256-thread block
    qkv_proj_wmma<<<1024, 256, 0, stream>>>(x, Wq, Wk, Wv, q, k, v);
    // one thread per (b,g,h,w): 1,048,576 threads
    local_attn_kernel<<<4096, 256, 0, stream>>>(q, k, v, rel_t, rel_f, out, attn);
}
